// Decoder_50981261803631
// MI455X (gfx1250) — matrix operand order repaired and checked by reading, not compiled, not run on hardware
//
#include <hip/hip_runtime.h>
#include <math.h>

// CDNA5 / gfx1250 decoder: attention + LSTM, all GEMM-like ops (including the
// attention score batch) on v_wmma_f32_16x16x32_f16. One wave32 per block,
// 16 batch rows per wave. B operands pre-swizzled in LDS (one 32B read/lane).
// Cell state lives in registers; outputs staged in LDS for coalesced b128 stores.

typedef __attribute__((ext_vector_type(16))) _Float16 v16h;
typedef __attribute__((ext_vector_type(8)))  float    v8f;

#define HID   32
#define HTT   8
#define FTT   3
#define ODIM  10
#define ROWS  16   // batch rows per block (one 16xK WMMA A-tile)

// B tile table: 0..1 Wk(n0=0,16) 2..3 Wv 4..5 Wq 6..13 Wih(n0=16j) 14..21 Whh 22 Wout
#define NT_WK   0
#define NT_WV   2
#define NT_WQ   4
#define NT_WIH  6
#define NT_WHH  14
#define NT_WOUT 22
#define NTILES  23

__device__ __forceinline__ v8f wmma(v16h a, v16h b, v8f c) {
  return __builtin_amdgcn_wmma_f32_16x16x32_f16(
      /*neg_a=*/false, a, /*neg_b=*/false, b,
      /*c_mod=*/(short)0, c, /*reuse_a=*/false, /*reuse_b=*/false);
}

__device__ __forceinline__ v8f splat8(float x) {
  v8f c;
#pragma unroll
  for (int r = 0; r < 8; ++r) c[r] = x;
  return c;
}

__device__ __forceinline__ float sigm(float x) { return 1.0f / (1.0f + __expf(-x)); }

// ---- B-tile staging: global f32 [N][32] row-major -> per-lane swizzled v16h.
// B 32x16 order used here (matches build_a): lane (n = lane&15, hi = lane>>4) holds K = hi*16 + s.
__device__ __forceinline__ void stage_b_tile(v16h* __restrict__ dstTile,
                                             const float* __restrict__ W,
                                             int n0, int lane, int nValid) {
  const int n  = (lane & 15) + n0;
  const int kb = ((lane >> 4) & 1) * 16;
  v16h t;
  if (n < nValid) {
    const float4* p = (const float4*)(W + n * HID + kb);
#pragma unroll
    for (int q = 0; q < 4; ++q) {
      const float4 x = p[q];
      t[4 * q + 0] = (_Float16)x.x;
      t[4 * q + 1] = (_Float16)x.y;
      t[4 * q + 2] = (_Float16)x.z;
      t[4 * q + 3] = (_Float16)x.w;
    }
  } else {
#pragma unroll
    for (int s = 0; s < 16; ++s) t[s] = (_Float16)0.0f;
  }
  dstTile[lane] = t;  // 32B aligned -> 2x ds_store_b128
}

// ---- A 16x32 f16 operand from LDS f16 row-major [16][stride].
// Lane (m = lane&15, hi): K = hi*16 + s.
__device__ __forceinline__ v16h build_a(const _Float16* __restrict__ src,
                                        int stride, int lane) {
  const int m = lane & 15, hb = ((lane >> 4) & 1) * 16;
  const _Float16* row = src + m * stride + hb;
  v16h a;
#pragma unroll
  for (int s = 0; s < 16; ++s) a[s] = row[s];
  return a;
}

// ---- A operand straight from global f32 [16][stride]: 4x global_load_b128 + cvt.
__device__ __forceinline__ v16h load_a_global(const float* __restrict__ src,
                                              int stride, int lane) {
  const int m = lane & 15, hb = ((lane >> 4) & 1) * 16;
  const float4* p0 = (const float4*)(src + m * stride + hb);
  const float4* p1 = (const float4*)(src + m * stride + hb + 8);
  const float4 x0 = p0[0], x1 = p0[1], y0 = p1[0], y1 = p1[1];
  v16h a;
  a[0]  = (_Float16)x0.x; a[1]  = (_Float16)x0.y; a[2]  = (_Float16)x0.z; a[3]  = (_Float16)x0.w;
  a[4]  = (_Float16)x1.x; a[5]  = (_Float16)x1.y; a[6]  = (_Float16)x1.z; a[7]  = (_Float16)x1.w;
  a[8]  = (_Float16)y0.x; a[9]  = (_Float16)y0.y; a[10] = (_Float16)y0.z; a[11] = (_Float16)y0.w;
  a[12] = (_Float16)y1.x; a[13] = (_Float16)y1.y; a[14] = (_Float16)y1.z; a[15] = (_Float16)y1.w;
  return a;
}

// ---- C 16x16 f32 scatter: VGPR r holds (m = r + hi*8, n = lane&15).
__device__ __forceinline__ void store_cf(float* __restrict__ dst, int stride,
                                         int n0, int lane, v8f c) {
  const int hi = (lane >> 4) & 1, n = (lane & 15) + n0;
#pragma unroll
  for (int r = 0; r < 8; ++r) dst[(r + hi * 8) * stride + n] = c[r];
}
__device__ __forceinline__ void store_c16(_Float16* __restrict__ dst, int stride,
                                          int n0, int lane, v8f c) {
  const int hi = (lane >> 4) & 1, n = (lane & 15) + n0;
#pragma unroll
  for (int r = 0; r < 8; ++r) dst[(r + hi * 8) * stride + n] = (_Float16)c[r];
}

__global__ void __launch_bounds__(32)
decoder_attn_lstm_kernel(const float* __restrict__ h_enc,
                         const float* __restrict__ Wq, const float* __restrict__ bq,
                         const float* __restrict__ Wk, const float* __restrict__ bk,
                         const float* __restrict__ Wv, const float* __restrict__ bv,
                         const float* __restrict__ W_ih, const float* __restrict__ b_ih,
                         const float* __restrict__ W_hh, const float* __restrict__ b_hh,
                         const float* __restrict__ W_out, const float* __restrict__ b_out,
                         float* __restrict__ out) {
  __shared__ v16h     sBt[NTILES][32];          // 23 KB swizzled B operands
  __shared__ float    sBq[HID], sBk[HID], sBv[HID], sBg[4 * HID], sBo[16];
  __shared__ _Float16 sK[ROWS * HTT * HID];     // 8 KB  f16: scores B operand
  __shared__ float    sV[ROWS * HTT * HID];     // 16 KB f32: context source
  __shared__ _Float16 sQ16[ROWS * HID];         // 1 KB  q (A operand for scores)
  __shared__ _Float16 sH16[ROWS * HID];         // 1 KB  hidden (A operand)
  __shared__ _Float16 sCtx16[ROWS * HID];       // 1 KB  context (A operand)
  __shared__ float    sAtt[ROWS * HTT];         // 0.5 KB
  __shared__ float    sLogit[ROWS * 16];        // 1 KB
  __shared__ __align__(16) float sOut[ROWS * FTT * ODIM];  // 1.9 KB staged output

  const int lane = threadIdx.x;
  const long long base = (long long)blockIdx.x * ROWS;

  // ---- stage swizzled B tiles ---------------------------------------------
  stage_b_tile(sBt[NT_WK + 0], Wk, 0,  lane, HID);
  stage_b_tile(sBt[NT_WK + 1], Wk, 16, lane, HID);
  stage_b_tile(sBt[NT_WV + 0], Wv, 0,  lane, HID);
  stage_b_tile(sBt[NT_WV + 1], Wv, 16, lane, HID);
  stage_b_tile(sBt[NT_WQ + 0], Wq, 0,  lane, HID);
  stage_b_tile(sBt[NT_WQ + 1], Wq, 16, lane, HID);
#pragma unroll
  for (int j = 0; j < 8; ++j) {
    stage_b_tile(sBt[NT_WIH + j], W_ih, j * 16, lane, 4 * HID);
    stage_b_tile(sBt[NT_WHH + j], W_hh, j * 16, lane, 4 * HID);
  }
  stage_b_tile(sBt[NT_WOUT], W_out, 0, lane, ODIM);

  if (lane < HID) { sBq[lane] = bq[lane]; sBk[lane] = bk[lane]; sBv[lane] = bv[lane]; }
  for (int i = lane; i < 4 * HID; i += 32) sBg[i] = b_ih[i] + b_hh[i];
  if (lane < 16) sBo[lane] = (lane < ODIM) ? b_out[lane] : 0.0f;
  for (int i = lane; i < ROWS * HID; i += 32) sH16[i] = (_Float16)0.0f;
  __syncthreads();

  // ---- K / V projections ---------------------------------------------------
  {
    const v16h bK0 = sBt[NT_WK + 0][lane], bK1 = sBt[NT_WK + 1][lane];
    const v16h bV0 = sBt[NT_WV + 0][lane], bV1 = sBt[NT_WV + 1][lane];
    const float bk0 = sBk[lane & 15], bk1 = sBk[(lane & 15) + 16];
    const float bv0 = sBv[lane & 15], bv1 = sBv[(lane & 15) + 16];
    const float* hbase = h_enc + base * (HTT * HID);
#pragma unroll
    for (int t = 0; t < HTT; ++t) {
      const v16h a = load_a_global(hbase + t * HID, HTT * HID, lane);
      v8f ck0 = splat8(bk0), ck1 = splat8(bk1);
      v8f cv0 = splat8(bv0), cv1 = splat8(bv1);
      ck0 = wmma(a, bK0, ck0);
      ck1 = wmma(a, bK1, ck1);
      cv0 = wmma(a, bV0, cv0);
      cv1 = wmma(a, bV1, cv1);
      store_c16(sK + t * HID, HTT * HID, 0,  lane, ck0);  // f16 (scores B op)
      store_c16(sK + t * HID, HTT * HID, 16, lane, ck1);
      store_cf (sV + t * HID, HTT * HID, 0,  lane, cv0);  // f32 (context src)
      store_cf (sV + t * HID, HTT * HID, 16, lane, cv1);
    }
  }
  __syncthreads();

  const float scale = 0.17677669529663687f;  // 1/sqrt(32)

  // LSTM cell state: this lane exclusively owns cells (m = r+hi*8, n = (lane&15)+nh*16)
  float cc[16];
#pragma unroll
  for (int i = 0; i < 16; ++i) cc[i] = 0.0f;

  for (int ft = 0; ft < FTT; ++ft) {
    // ---- Q = h @ Wq^T + bq (f16, consumed as scores A operand) -------------
    {
      const v16h ah = build_a(sH16, HID, lane);
      v8f c0 = splat8(sBq[lane & 15]);
      v8f c1 = splat8(sBq[(lane & 15) + 16]);
      c0 = wmma(ah, sBt[NT_WQ + 0][lane], c0);
      c1 = wmma(ah, sBt[NT_WQ + 1][lane], c1);
      store_c16(sQ16, HID, 0,  lane, c0);
      store_c16(sQ16, HID, 16, lane, c1);
    }
    __syncthreads();

    // ---- attention scores via WMMA ----------------------------------------
    // Tile j covers columns c = 16j+n, c -> (m = c>>3, t = c&7). B column's
    // 16 K-dim halfs are contiguous in sK. Useful (block-diagonal) entry of
    // C sits at register r = 2*(j&3) + (n>>3) in half hi == (j>=4).
    {
      const v16h aq = build_a(sQ16, HID, lane);
      const int n = lane & 15, hi = (lane >> 4) & 1;
      const int nh3 = n >> 3;      // 0/1
      const int tcol = n & 7;      // t of this lane's column
#pragma unroll
      for (int j = 0; j < 8; ++j) {
        const int mc = 2 * j + nh3;
        const _Float16* kp = sK + mc * (HTT * HID) + tcol * HID + hi * 16;
        v16h bk;
#pragma unroll
        for (int s = 0; s < 16; ++s) bk[s] = kp[s];   // 2x ds_load_b128
        v8f cs = splat8(0.0f);
        cs = wmma(aq, bk, cs);
        const float val = (nh3 ? cs[2 * (j & 3) + 1] : cs[2 * (j & 3)]) * scale;
        if (hi == ((j >= 4) ? 1 : 0)) sAtt[16 * j + n] = val;
      }
    }
    __syncthreads();

    // ---- softmax over 8 steps: 16 lanes ------------------------------------
    if (lane < ROWS) {
      const int m = lane;
      float mx = -1e30f;
#pragma unroll
      for (int t = 0; t < HTT; ++t) mx = fmaxf(mx, sAtt[m * HTT + t]);
      float e[HTT], sum = 0.0f;
#pragma unroll
      for (int t = 0; t < HTT; ++t) { e[t] = __expf(sAtt[m * HTT + t] - mx); sum += e[t]; }
      const float inv = 1.0f / sum;
#pragma unroll
      for (int t = 0; t < HTT; ++t) sAtt[m * HTT + t] = e[t] * inv;
    }
    __syncthreads();

    // ---- context: all 32 lanes (row m, d-half), float4 LDS reads -----------
    {
      const int m = lane & 15, d0 = ((lane >> 4) & 1) * 16;
      float w[HTT];
#pragma unroll
      for (int t = 0; t < HTT; ++t) w[t] = sAtt[m * HTT + t];
      float acc[16];
#pragma unroll
      for (int dd = 0; dd < 16; ++dd) acc[dd] = 0.0f;
#pragma unroll
      for (int t = 0; t < HTT; ++t) {
        const float4* vp = (const float4*)(sV + m * (HTT * HID) + t * HID + d0);
#pragma unroll
        for (int q4 = 0; q4 < 4; ++q4) {
          const float4 x = vp[q4];
          acc[4 * q4 + 0] += w[t] * x.x;
          acc[4 * q4 + 1] += w[t] * x.y;
          acc[4 * q4 + 2] += w[t] * x.z;
          acc[4 * q4 + 3] += w[t] * x.w;
        }
      }
      v16h pk;
#pragma unroll
      for (int dd = 0; dd < 16; ++dd) pk[dd] = (_Float16)acc[dd];
      *(v16h*)(sCtx16 + m * HID + d0) = pk;  // 32B aligned packed store
    }
    __syncthreads();

    // ---- LSTM gates: ctx @ Wih^T + h @ Whh^T + (b_ih+b_hh) ----------------
    {
      const v16h ax = build_a(sCtx16, HID, lane);
      const v16h ah = build_a(sH16, HID, lane);
      const int hi = (lane >> 4) & 1;
#pragma unroll
      for (int nh = 0; nh < 2; ++nh) {
        const int nloc = (lane & 15) + nh * 16;   // column within each gate
        v8f ci = splat8(sBg[0 * HID + nloc]);
        v8f cf = splat8(sBg[1 * HID + nloc]);
        v8f cg = splat8(sBg[2 * HID + nloc]);
        v8f co = splat8(sBg[3 * HID + nloc]);
        ci = wmma(ax, sBt[NT_WIH + 0 + nh][lane], ci);
        ci = wmma(ah, sBt[NT_WHH + 0 + nh][lane], ci);
        cf = wmma(ax, sBt[NT_WIH + 2 + nh][lane], cf);
        cf = wmma(ah, sBt[NT_WHH + 2 + nh][lane], cf);
        cg = wmma(ax, sBt[NT_WIH + 4 + nh][lane], cg);
        cg = wmma(ah, sBt[NT_WHH + 4 + nh][lane], cg);
        co = wmma(ax, sBt[NT_WIH + 6 + nh][lane], co);
        co = wmma(ah, sBt[NT_WHH + 6 + nh][lane], co);
        // activations applied directly on C registers (m = r + hi*8, n = nloc)
#pragma unroll
        for (int r = 0; r < 8; ++r) {
          const int m = r + hi * 8;
          const float i_ = sigm(ci[r]);
          const float f_ = sigm(cf[r]);
          const float g_ = tanhf(cg[r]);
          const float o_ = sigm(co[r]);
          const float cN = f_ * cc[nh * 8 + r] + i_ * g_;
          const float hN = o_ * tanhf(cN);
          cc[nh * 8 + r] = cN;
          sH16[m * HID + nloc] = (_Float16)hN;
        }
      }
    }
    __syncthreads();

    // ---- logits = h @ Wout^T + b_out (N padded to 16) ----------------------
    {
      const v16h ah = build_a(sH16, HID, lane);
      v8f c = splat8(sBo[lane & 15]);
      c = wmma(ah, sBt[NT_WOUT][lane], c);
      store_cf(sLogit, 16, 0, lane, c);
    }
    __syncthreads();

    // ---- log-softmax over 10 classes -> staged output ----------------------
    if (lane < ROWS) {
      const int m = lane;
      float mx = -1e30f;
      for (int j = 0; j < ODIM; ++j) mx = fmaxf(mx, sLogit[m * 16 + j]);
      float sum = 0.0f;
      for (int j = 0; j < ODIM; ++j) sum += __expf(sLogit[m * 16 + j] - mx);
      const float ls = mx + __logf(sum);
      for (int j = 0; j < ODIM; ++j)
        sOut[m * (FTT * ODIM) + ft * ODIM + j] = sLogit[m * 16 + j] - ls;
    }
    __syncthreads();
  }

  // ---- coalesced output: block's slab is 480 floats = 120 float4 ----------
  {
    float4* op = (float4*)(out + base * (FTT * ODIM));
    const float4* sp = (const float4*)sOut;
    for (int i = lane; i < (ROWS * FTT * ODIM) / 4; i += 32)
      op[i] = sp[i];   // ds_load_b128 + global_store_b128
  }
}

extern "C" void kernel_launch(void* const* d_in, const int* in_sizes, int n_in,
                              void* d_out, int out_size, void* d_ws, size_t ws_size,
                              hipStream_t stream) {
  const float* h_enc = (const float*)d_in[0];
  const float* Wq    = (const float*)d_in[1];
  const float* bq    = (const float*)d_in[2];
  const float* Wk    = (const float*)d_in[3];
  const float* bk    = (const float*)d_in[4];
  const float* Wv    = (const float*)d_in[5];
  const float* bv    = (const float*)d_in[6];
  const float* W_ih  = (const float*)d_in[7];
  const float* b_ih  = (const float*)d_in[8];
  const float* W_hh  = (const float*)d_in[9];
  const float* b_hh  = (const float*)d_in[10];
  const float* W_out = (const float*)d_in[11];
  const float* b_out = (const float*)d_in[12];
  float* out = (float*)d_out;

  const int nrows  = in_sizes[0] / (HTT * HID);  // B
  const int blocks = nrows / ROWS;               // 16384 for B=262144

  decoder_attn_lstm_kernel<<<blocks, 32, 0, stream>>>(
      h_enc, Wq, bq, Wk, bk, Wv, bv, W_ih, b_ih, W_hh, b_hh, W_out, b_out, out);
}
